// MultiHeadAttention_60284160966755
// MI455X (gfx1250) — compile-verified, matrix operand order checked by reading, not run on hardware
//
#include <hip/hip_runtime.h>
#include <hip/hip_bf16.h>

// ---------------------------------------------------------------------------
// GATv2 multi-head attention: N=50000 nodes, E=1.6M edges, IN_DIM=128, H=4, D=32
// Phase 0: pre-swizzle x and W_{src,dst,val} into bf16 WMMA operand layouts
// Phase 1: bf16 WMMA projections (x @ W_src/W_dst/W_val), f32 accumulate
// Phase 2: edge scores + segment-max (fused, atomic uint-ordered max)
// Phase 3: exp + segment-sum
// Phase 4: normalize -> attn output + weighted scatter-add of h_v
// ---------------------------------------------------------------------------

typedef __attribute__((ext_vector_type(16))) __bf16 v16bf;
typedef __attribute__((ext_vector_type(8)))  float  v8f;

#define HD 128      // H*D and IN_DIM (both 128)
#define HEADS 4
#define DDIM 32
#define LEAKY 0.2f

// float -> bf16, round to nearest even
__device__ __forceinline__ __bf16 f2bf(float f) {
    unsigned u = __float_as_uint(f);
    unsigned r = (u + 0x7FFFu + ((u >> 16) & 1u)) >> 16;
    return __builtin_bit_cast(__bf16, (unsigned short)r);
}

// K-within-tile index for the ISA 16-bit operand layout:
// halves 2v,2v+1 of the wave's 16-half operand hold K = base(v) + koff + ...
__device__ __forceinline__ int klocal(int half_idx, int lane) {
    const int vv   = half_idx >> 1;
    const int koff = (lane >> 4) * 8;      // upper half-wave K offset
    return ((vv < 4) ? 0 : 16) + koff + (vv & 3) * 2 + (half_idx & 1);
}

// order-preserving float<->uint mapping for atomic max on signed floats
__device__ __forceinline__ unsigned encf(float f) {
    unsigned u = __float_as_uint(f);
    return (u & 0x80000000u) ? ~u : (u | 0x80000000u);
}
__device__ __forceinline__ float decf(unsigned u) {
    return (u & 0x80000000u) ? __uint_as_float(u ^ 0x80000000u)
                             : __uint_as_float(~u);
}

// ---------------------------------------------------------------------------
// Phase 0a: pack x into A-operand layout, bf16.
// A_bf[((rb*4 + kt)*32 + lane)*16 + i] = bf16(x[(rb*16 + lane&15)*128 + kt*32 + klocal(i,lane)])
// ---------------------------------------------------------------------------
__global__ void __launch_bounds__(256)
pack_A_kernel(const float* __restrict__ x, __bf16* __restrict__ A_bf,
              int n_nodes, int n_blk)
{
    const int idx = blockIdx.x * blockDim.x + threadIdx.x;
    if (idx >= n_blk * 2048) return;               // 4 kt * 32 lanes * 16 halves
    const int i    = idx & 15;
    const int lane = (idx >> 4) & 31;
    const int kt   = (idx >> 9) & 3;
    const int rb   = idx >> 11;
    const int row  = rb * 16 + (lane & 15);
    const int k    = kt * 32 + klocal(i, lane);
    const float v  = (row < n_nodes) ? x[(size_t)row * HD + k] : 0.f;
    A_bf[idx] = f2bf(v);
}

// ---------------------------------------------------------------------------
// Phase 0b: pack the three W matrices into B-operand layout, bf16.
// B_bf[(((w*4 + kt)*8 + n)*32 + lane)*16 + i] = bf16(W_w[(kt*32 + klocal(i,lane))*128 + n*16 + lane&15])
// ---------------------------------------------------------------------------
__global__ void __launch_bounds__(256)
pack_B_kernel(const float* __restrict__ Wsrc,
              const float* __restrict__ Wdst,
              const float* __restrict__ Wval,
              __bf16* __restrict__ B_bf)
{
    const int idx = blockIdx.x * blockDim.x + threadIdx.x;
    if (idx >= 3 * 4 * 8 * 32 * 16) return;        // 49152
    const int i    = idx & 15;
    const int lane = (idx >> 4) & 31;
    const int n    = (idx >> 9) & 7;
    const int kt   = (idx >> 12) & 3;
    const int w    = idx >> 14;
    const float* W = (w == 0) ? Wsrc : (w == 1) ? Wdst : Wval;
    const int k    = kt * 32 + klocal(i, lane);
    const int ncol = n * 16 + (lane & 15);
    B_bf[idx] = f2bf(W[(size_t)k * HD + ncol]);
}

// ---------------------------------------------------------------------------
// Phase 1: projection GEMM.  Block = 96 threads = 3 waves; wave w computes a
// 16x128 output slab of x @ W_w for its row block.  Operands are pre-swizzled:
// per kt one 32B A load, per (kt,n) one 32B B load, then WMMA.
// ---------------------------------------------------------------------------
__global__ void __launch_bounds__(96)
proj_gemm_kernel(const __bf16* __restrict__ A_bf,
                 const __bf16* __restrict__ B_bf,
                 float* __restrict__ hs,
                 float* __restrict__ hd,
                 float* __restrict__ hv,
                 int n_nodes)
{
    const int lane  = threadIdx.x & 31;
    const int matId = threadIdx.x >> 5;           // 0,1,2 -> src,dst,val
    float* out      = (matId == 0) ? hs : (matId == 1) ? hd : hv;

    const int rb = blockIdx.x;

    v8f acc[8];
    #pragma unroll
    for (int n = 0; n < 8; ++n) acc[n] = v8f{0.f,0.f,0.f,0.f,0.f,0.f,0.f,0.f};

    #pragma unroll
    for (int kt = 0; kt < 4; ++kt) {              // K = 128 in 4 chunks of 32
        const v16bf a = *(const v16bf*)(A_bf + ((size_t)(rb * 4 + kt) * 32 + lane) * 16);
        #pragma unroll
        for (int n = 0; n < 8; ++n) {             // 8 column tiles of 16
            const v16bf b = *(const v16bf*)(B_bf +
                ((size_t)((matId * 4 + kt) * 8 + n) * 32 + lane) * 16);
            acc[n] = __builtin_amdgcn_wmma_f32_16x16x32_bf16(
                false, a, false, b, (short)0, acc[n], false, false);
        }
    }

    // C/D layout: VGPR rr -> row rr (+8 for upper half-wave), col = lane&15
    const int rows = rb * 16;
    #pragma unroll
    for (int n = 0; n < 8; ++n) {
        union { v8f v; float e[8]; } c; c.v = acc[n];
        const int col = n * 16 + (lane & 15);
        #pragma unroll
        for (int rr = 0; rr < 8; ++rr) {
            const int row = rows + rr + ((lane >> 4) * 8);
            if (row < n_nodes) out[(size_t)row * HD + col] = c.e[rr];
        }
    }
}

// ---------------------------------------------------------------------------
// Init: zero out_sum, seg_sum, seg_max (encoded)
// ---------------------------------------------------------------------------
__global__ void init_kernel(float* __restrict__ out_sum,
                            float* __restrict__ seg_sum,
                            unsigned* __restrict__ seg_max,
                            int n_out, int n_seg)
{
    const int stride = gridDim.x * blockDim.x;
    for (int i = blockIdx.x * blockDim.x + threadIdx.x; i < n_out; i += stride)
        out_sum[i] = 0.f;
    for (int i = blockIdx.x * blockDim.x + threadIdx.x; i < n_seg; i += stride) {
        seg_sum[i] = 0.f;
        seg_max[i] = 0u;               // decodes below every real score
    }
}

// ---------------------------------------------------------------------------
// Phase 2: one wave per edge.  Lane l owns channels [4l,4l+4); head = l/8.
// score[e,h] = sum_d a[h,d] * leaky_relu(hs[src,h,d] + hd[dst,h,d])
// Fused atomic segment-max per (dst, head).
// ---------------------------------------------------------------------------
__global__ void __launch_bounds__(256)
edge_score_kernel(const float* __restrict__ hs,
                  const float* __restrict__ hd,
                  const float* __restrict__ av,     // attn_vec, 128 floats
                  const int* __restrict__ src,
                  const int* __restrict__ dst,
                  float* __restrict__ scores,       // [E, H]
                  unsigned* __restrict__ seg_max,   // [N, H] encoded
                  int n_edges)
{
    const int e = blockIdx.x * 8 + (threadIdx.x >> 5);
    if (e >= n_edges) return;                      // wave-uniform
    const int lane = threadIdx.x & 31;
    const int s = src[e], d = dst[e];

    const float4 vs = *(const float4*)(hs + (size_t)s * HD + lane * 4);
    const float4 vd = *(const float4*)(hd + (size_t)d * HD + lane * 4);
    const float4 va = *(const float4*)(av + lane * 4);

    float t, p = 0.f;
    t = vs.x + vd.x; t = (t > 0.f) ? t : LEAKY * t; p += t * va.x;
    t = vs.y + vd.y; t = (t > 0.f) ? t : LEAKY * t; p += t * va.y;
    t = vs.z + vd.z; t = (t > 0.f) ? t : LEAKY * t; p += t * va.z;
    t = vs.w + vd.w; t = (t > 0.f) ? t : LEAKY * t; p += t * va.w;

    // reduce over the 8 lanes of each head (lanes h*8 .. h*8+7)
    p += __shfl_xor(p, 1);
    p += __shfl_xor(p, 2);
    p += __shfl_xor(p, 4);

    const int h = lane >> 3;
    if ((lane & 7) == 0) {
        scores[(size_t)e * HEADS + h] = p;
        atomicMax(&seg_max[(size_t)d * HEADS + h], encf(p));
    }
}

// ---------------------------------------------------------------------------
// Phase 3: exp(score - seg_max[dst]) in place + atomic segment-sum
// ---------------------------------------------------------------------------
__global__ void __launch_bounds__(256)
exp_sum_kernel(float* __restrict__ scores,          // [E,H] in/out -> exp
               const int* __restrict__ dst,
               const unsigned* __restrict__ seg_max,
               float* __restrict__ seg_sum,
               int n_edges)
{
    const int idx = blockIdx.x * blockDim.x + threadIdx.x;
    if (idx >= n_edges * HEADS) return;
    const int e = idx >> 2;                        // HEADS == 4
    const int h = idx & 3;
    const int d = dst[e];
    const float m  = decf(seg_max[(size_t)d * HEADS + h]);
    const float ex = expf(scores[idx] - m);
    scores[idx] = ex;
    atomicAdd(&seg_sum[(size_t)d * HEADS + h], ex);
}

// ---------------------------------------------------------------------------
// Phase 4: one wave per edge.  attn = exp / (sum + 1e-9); write attn output
// and scatter-add attn * h_v[src] into out_sum[dst].
// ---------------------------------------------------------------------------
__global__ void __launch_bounds__(256)
aggregate_kernel(const float* __restrict__ exp_scores,  // [E,H]
                 const float* __restrict__ hv,
                 const int* __restrict__ src,
                 const int* __restrict__ dst,
                 const float* __restrict__ seg_sum,
                 float* __restrict__ out_sum,            // [N, H*D]
                 float* __restrict__ attn_out,           // [E, H]
                 int n_edges)
{
    const int e = blockIdx.x * 8 + (threadIdx.x >> 5);
    if (e >= n_edges) return;                      // wave-uniform
    const int lane = threadIdx.x & 31;
    const int s = src[e], d = dst[e];
    const int h = lane >> 3;

    const float ex  = exp_scores[(size_t)e * HEADS + h];
    const float sm  = seg_sum[(size_t)d * HEADS + h];
    const float atn = ex / (sm + 1e-9f);

    if ((lane & 7) == 0) attn_out[(size_t)e * HEADS + h] = atn;

    const float4 v = *(const float4*)(hv + (size_t)s * HD + lane * 4);
    float* o = out_sum + (size_t)d * HD + lane * 4;
    atomicAdd(o + 0, atn * v.x);
    atomicAdd(o + 1, atn * v.y);
    atomicAdd(o + 2, atn * v.z);
    atomicAdd(o + 3, atn * v.w);
}

// ---------------------------------------------------------------------------
extern "C" void kernel_launch(void* const* d_in, const int* in_sizes, int n_in,
                              void* d_out, int out_size, void* d_ws, size_t ws_size,
                              hipStream_t stream)
{
    const float* x    = (const float*)d_in[0];   // [N, 128]
    const int*   ei   = (const int*)d_in[1];     // [2, E]
    const float* Wsrc = (const float*)d_in[2];   // [128, 128]
    const float* Wdst = (const float*)d_in[3];
    const float* Wval = (const float*)d_in[4];
    const float* av   = (const float*)d_in[5];   // [1, H, D] = 128 floats

    const int N = in_sizes[0] / HD;
    const int E = in_sizes[1] / 2;
    const int* src = ei;
    const int* dst = ei + E;

    float* out_sum  = (float*)d_out;                      // [N, 128]
    float* attn_out = out_sum + (size_t)N * HD;           // [E, 4]

    // workspace layout (all sub-buffers stay 32B aligned for vector loads)
    float*    ws      = (float*)d_ws;
    const size_t nhd  = (size_t)N * HD;
    float*    hs      = ws;
    float*    hdp     = hs  + nhd;
    float*    hv      = hdp + nhd;
    float*    scores  = hv  + nhd;                        // [E,H], reused as exp
    float*    seg_sum = scores + (size_t)E * HEADS;       // [N,H]
    unsigned* seg_max = (unsigned*)(seg_sum + (size_t)N * HEADS);
    const int n_blk   = (N + 15) / 16;
    __bf16*   A_bf    = (__bf16*)(seg_max + (size_t)N * HEADS);   // [n_blk*2048]
    __bf16*   B_bf    = A_bf + (size_t)n_blk * 2048;              // [49152]

    // zero accumulators
    init_kernel<<<1024, 256, 0, stream>>>(out_sum, seg_sum, seg_max,
                                          N * HD, N * HEADS);

    // pre-swizzle operands into WMMA layouts (bf16)
    pack_A_kernel<<<(n_blk * 2048 + 255) / 256, 256, 0, stream>>>(x, A_bf, N, n_blk);
    pack_B_kernel<<<(49152 + 255) / 256, 256, 0, stream>>>(Wsrc, Wdst, Wval, B_bf);

    // projections: one 3-wave block per 16-row slab
    proj_gemm_kernel<<<n_blk, 96, 0, stream>>>(A_bf, B_bf, hs, hdp, hv, N);

    // edge scores + segment max (8 edges / 256-thread block, 1 wave per edge)
    edge_score_kernel<<<(E + 7) / 8, 256, 0, stream>>>(
        hs, hdp, av, src, dst, scores, seg_max, E);

    // exp + segment sum
    exp_sum_kernel<<<(E * HEADS + 255) / 256, 256, 0, stream>>>(
        scores, dst, seg_max, seg_sum, E);

    // normalize + weighted scatter-add
    aggregate_kernel<<<(E + 7) / 8, 256, 0, stream>>>(
        scores, hv, src, dst, seg_sum, out_sum, attn_out, E);
}
